// MainBlockParallel_12936441495993
// MI455X (gfx1250) — matrix operand order repaired and checked by reading, not compiled, not run on hardware
//
#include <hip/hip_runtime.h>

typedef __attribute__((ext_vector_type(16))) __bf16 v16bf;
typedef __attribute__((ext_vector_type(8)))  __bf16 v8bf;
typedef __attribute__((ext_vector_type(8)))  float  v8f;

#define BB 2
#define NN 512
#define MM 77
#define DIMV 512
#define CTXD 768
#define DINNER 1024
#define DSTATE 16

// ---------------------------------------------------------------------------
// bf16 WMMA GEMM. A: row-major bf16 [Mpad x K] (lda), rows zero-padded to 32.
// Bt: bf16 weights transposed to [N x K] (K contiguous). One wave computes a
// 32 x (16*NT) tile: 2 A fragments x NT B fragments -> 2*NT WMMAs per K-slab
// (1.5 b128 loads per WMMA at NT=4). All fragment loads are issued before the
// WMMA group so loadcnt waits are incremental (load/tensor overlap).
// Fragment layouts per CDNA5 ISA 7.12.2:
//   A frag = rows mrow;  elements [k0+half*8,+8) ++ [k0+16+half*8,+8)
//   B frag = col ncol;   elements [k0+half*8,+8) ++ [k0+16+half*8,+8)
//   C/D    = col ncol, VGPR r -> row base + half*8 + r
// EPI: 0 none, 1 +bias, 2 +bias,softplus.  OUT: 0 f32, 1 bf16, 2 both.
// ---------------------------------------------------------------------------
template<int EPI, int NT, int OUT>
__global__ __launch_bounds__(128)
void gemm_bf16_kernel(const __bf16* __restrict__ Abf, int lda,
                      const __bf16* __restrict__ Bt,
                      const float* __restrict__ bias,
                      float* __restrict__ C, __bf16* __restrict__ Cbf,
                      int ldc, int M, int N, int K)
{
    int wid  = (blockIdx.x * blockDim.x + threadIdx.x) >> 5;
    int lane = threadIdx.x & 31;
    int tilesN = N / (16 * NT);
    int tilesM = (M + 31) >> 5;
    if (wid >= tilesM * tilesN) return;
    int tm = wid / tilesN, tn = wid % tilesN;
    int half = lane >> 4;
    int l16  = lane & 15;

    const __bf16* ap0 = Abf + (size_t)(tm * 32 + l16) * lda + half * 8;
    const __bf16* ap1 = ap0 + (size_t)16 * lda;
    const __bf16* bp[NT];
    #pragma unroll
    for (int t = 0; t < NT; ++t)
        bp[t] = Bt + (size_t)(tn * 16 * NT + t * 16 + l16) * K + half * 8;

    v8f acc[2][NT];
    #pragma unroll
    for (int mi = 0; mi < 2; ++mi)
        #pragma unroll
        for (int t = 0; t < NT; ++t)
            acc[mi][t] = (v8f){0.f,0.f,0.f,0.f,0.f,0.f,0.f,0.f};

    for (int k0 = 0; k0 < K; k0 += 32) {
        // ---- issue ALL fragment loads first ----
        v8bf a0lo = *(const v8bf*)(ap0 + k0);
        v8bf a0hi = *(const v8bf*)(ap0 + k0 + 16);
        v8bf a1lo = *(const v8bf*)(ap1 + k0);
        v8bf a1hi = *(const v8bf*)(ap1 + k0 + 16);
        v16bf bfrag[NT];
        #pragma unroll
        for (int t = 0; t < NT; ++t) {
            v8bf blo = *(const v8bf*)(bp[t] + k0);
            v8bf bhi = *(const v8bf*)(bp[t] + k0 + 16);
            bfrag[t] = __builtin_shufflevector(blo, bhi,
                           0,1,2,3,4,5,6,7,8,9,10,11,12,13,14,15);
        }
        if (k0 + 32 < K) {
            __builtin_prefetch(ap0 + k0 + 32, 0, 3);
            __builtin_prefetch(ap1 + k0 + 32, 0, 3);
        }
        v16bf af0 = __builtin_shufflevector(a0lo, a0hi,
                        0,1,2,3,4,5,6,7,8,9,10,11,12,13,14,15);
        v16bf af1 = __builtin_shufflevector(a1lo, a1hi,
                        0,1,2,3,4,5,6,7,8,9,10,11,12,13,14,15);
        // ---- WMMA group (independent accumulators, incremental waits) ----
        #pragma unroll
        for (int t = 0; t < NT; ++t) {
            acc[0][t] = __builtin_amdgcn_wmma_f32_16x16x32_bf16(
                            false, af0, false, bfrag[t], (short)0, acc[0][t], false, false);
            acc[1][t] = __builtin_amdgcn_wmma_f32_16x16x32_bf16(
                            false, af1, false, bfrag[t], (short)0, acc[1][t], false, false);
        }
    }

    #pragma unroll
    for (int mi = 0; mi < 2; ++mi) {
        int mbase = tm * 32 + mi * 16 + half * 8;
        #pragma unroll
        for (int t = 0; t < NT; ++t) {
            int ncol = tn * 16 * NT + t * 16 + l16;
            #pragma unroll
            for (int r = 0; r < 8; ++r) {
                int row = mbase + r;
                if (row < M) {
                    float v = acc[mi][t][r];
                    if (EPI >= 1) v += bias[ncol];
                    if (EPI == 2) v = (v > 20.f) ? v : log1pf(expf(v));
                    if constexpr (OUT == 0 || OUT == 2) C[(size_t)row * ldc + ncol] = v;
                    if constexpr (OUT >= 1) Cbf[(size_t)row * ldc + ncol] = (__bf16)v;
                }
            }
        }
    }
}

// weights: W[K x N] f32 -> Wt[N x K] bf16
__global__ void wt_cvt_kernel(const float* __restrict__ W, __bf16* __restrict__ Wt,
                              int K, int N)
{
    int tid = blockIdx.x * blockDim.x + threadIdx.x;
    if (tid >= K * N) return;
    int k = tid / N, n = tid - k * N;
    Wt[(size_t)n * K + k] = (__bf16)W[tid];
}

// activations: f32 -> bf16 row-major, rows zero-padded to rows_out
__global__ void cvt_pad_kernel(const float* __restrict__ in, __bf16* __restrict__ out,
                               int rows_in, int rows_out, int cols)
{
    int tid = blockIdx.x * blockDim.x + threadIdx.x;
    if (tid >= rows_out * cols) return;
    int r = tid / cols;
    out[tid] = (r < rows_in) ? (__bf16)in[tid] : (__bf16)0.f;
}

// ---------------------------------------------------------------------------
__global__ void ctx_mean_silu_kernel(const float* __restrict__ ctx, float* __restrict__ smean)
{
    int tid = blockIdx.x * blockDim.x + threadIdx.x;
    if (tid >= BB * CTXD) return;
    int b = tid / CTXD, c = tid % CTXD;
    float s = 0.f;
    for (int m = 0; m < MM; ++m) s += ctx[(size_t)(b * MM + m) * CTXD + c];
    float mu = s * (1.f / (float)MM);
    smean[tid] = mu / (1.f + expf(-mu));
}

__global__ void ss_proj_kernel(const float* __restrict__ smean, const float* __restrict__ W,
                               const float* __restrict__ bias, float* __restrict__ out)
{
    int tid = blockIdx.x * blockDim.x + threadIdx.x;
    if (tid >= BB * 1024) return;
    int b = tid >> 10, j = tid & 1023;
    const float* m = smean + b * CTXD;
    float acc = bias[j];
    for (int c = 0; c < CTXD; ++c) acc += m[c] * W[c * 1024 + j];
    out[tid] = acc;
}

// ---------------------------------------------------------------------------
// Attention: one wave per (b,h,n); writes bf16 (feeds Wo GEMM as A operand)
// ---------------------------------------------------------------------------
__global__ __launch_bounds__(128)
void attention_kernel(const float* __restrict__ q, const float* __restrict__ k,
                      const float* __restrict__ v, __bf16* __restrict__ out)
{
    int wid  = (blockIdx.x * blockDim.x + threadIdx.x) >> 5;
    int lane = threadIdx.x & 31;
    if (wid >= BB * 8 * NN) return;
    int b = wid >> 12;
    int h = (wid >> 9) & 7;
    int n = wid & 511;

    const float* qrow = q + (size_t)(b * NN + n) * 512 + h * 64;
    float q0 = qrow[lane], q1 = qrow[lane + 32];

    float sc[3] = {0.f, 0.f, 0.f};
    for (int mi = 0; mi < MM; ++mi) {
        const float* krow = k + (size_t)(b * MM + mi) * 512 + h * 64;
        float p = q0 * krow[lane] + q1 * krow[lane + 32];
        #pragma unroll
        for (int m = 16; m >= 1; m >>= 1) p += __shfl_xor(p, m);
        if ((mi & 31) == lane) sc[mi >> 5] = p * 0.125f;   // SCALE = 64^-0.5
    }
    float mx = -3.0e38f;
    #pragma unroll
    for (int j = 0; j < 3; ++j) if (j * 32 + lane < MM) mx = fmaxf(mx, sc[j]);
    #pragma unroll
    for (int m = 16; m >= 1; m >>= 1) mx = fmaxf(mx, __shfl_xor(mx, m));
    float es = 0.f;
    #pragma unroll
    for (int j = 0; j < 3; ++j) if (j * 32 + lane < MM) es += expf(sc[j] - mx);
    #pragma unroll
    for (int m = 16; m >= 1; m >>= 1) es += __shfl_xor(es, m);
    float inv = 1.f / es;

    float a0 = 0.f, a1 = 0.f;
    for (int mi = 0; mi < MM; ++mi) {
        float sv = __shfl(sc[mi >> 5], mi & 31);
        float w  = expf(sv - mx) * inv;
        const float* vrow = v + (size_t)(b * MM + mi) * 512 + h * 64;
        a0 += w * vrow[lane];
        a1 += w * vrow[lane + 32];
    }
    __bf16* orow = out + (size_t)(b * NN + n) * 512 + h * 64;
    orow[lane] = (__bf16)a0; orow[lane + 32] = (__bf16)a1;
}

// ---------------------------------------------------------------------------
// Row layer-norm (cols=512) -> bf16 (feeds in_proj GEMM)
// ---------------------------------------------------------------------------
__global__ __launch_bounds__(128)
void layernorm_rows_kernel(const float* __restrict__ in, const float* __restrict__ w,
                           const float* __restrict__ b, __bf16* __restrict__ out, int rows)
{
    int wid  = (blockIdx.x * blockDim.x + threadIdx.x) >> 5;
    int lane = threadIdx.x & 31;
    if (wid >= rows) return;
    const float* row = in + (size_t)wid * DIMV;
    float v[16]; float sum = 0.f, sq = 0.f;
    #pragma unroll
    for (int i = 0; i < 16; ++i) { float t = row[lane + i * 32]; v[i] = t; sum += t; sq += t * t; }
    #pragma unroll
    for (int m = 16; m >= 1; m >>= 1) { sum += __shfl_xor(sum, m); sq += __shfl_xor(sq, m); }
    float mu = sum * (1.f / DIMV);
    float var = sq * (1.f / DIMV) - mu * mu;
    float rs = rsqrtf(var + 1e-5f);
    __bf16* orow = out + (size_t)wid * DIMV;
    #pragma unroll
    for (int i = 0; i < 16; ++i) { int c = lane + i * 32; orow[c] = (__bf16)((v[i] - mu) * rs * w[c] + b[c]); }
}

// ---------------------------------------------------------------------------
// scale_shift -> layer_norm -> * scale ; optionally + residual + addbuf
// ---------------------------------------------------------------------------
__global__ __launch_bounds__(128)
void post_ss_ln_kernel(const float* __restrict__ in, const float* __restrict__ ss,
                       const float* __restrict__ nw, const float* __restrict__ nb,
                       const float* __restrict__ scale,
                       const float* __restrict__ resid, const float* __restrict__ addbuf,
                       float* __restrict__ out, int rows)
{
    int wid  = (blockIdx.x * blockDim.x + threadIdx.x) >> 5;
    int lane = threadIdx.x & 31;
    if (wid >= rows) return;
    int b = wid >> 9;
    const float* row = in + (size_t)wid * DIMV;
    const float* sc  = ss + b * 1024;
    const float* sh  = sc + DIMV;
    float v[16]; float sum = 0.f, sq = 0.f;
    #pragma unroll
    for (int i = 0; i < 16; ++i) {
        int c = lane + i * 32;
        float t = row[c] * (sc[c] + 1.f) + sh[c];
        v[i] = t; sum += t; sq += t * t;
    }
    #pragma unroll
    for (int m = 16; m >= 1; m >>= 1) { sum += __shfl_xor(sum, m); sq += __shfl_xor(sq, m); }
    float mu = sum * (1.f / DIMV);
    float var = sq * (1.f / DIMV) - mu * mu;
    float rs = rsqrtf(var + 1e-5f);
    float s  = scale[0];
    float* orow = out + (size_t)wid * DIMV;
    #pragma unroll
    for (int i = 0; i < 16; ++i) {
        int c = lane + i * 32;
        float r = ((v[i] - mu) * rs * nw[c] + nb[c]) * s;
        if (resid) r += resid[(size_t)wid * DIMV + c] + addbuf[(size_t)wid * DIMV + c];
        orow[c] = r;
    }
}

// ---------------------------------------------------------------------------
// depthwise causal conv (D_CONV=4) + silu; writes f32 (scan) and bf16 (GEMM A)
// ---------------------------------------------------------------------------
__global__ void conv_silu_kernel(const float* __restrict__ xz, const float* __restrict__ cw,
                                 const float* __restrict__ cb,
                                 float* __restrict__ xc, __bf16* __restrict__ xcb)
{
    int tid = blockIdx.x * blockDim.x + threadIdx.x;
    if (tid >= BB * NN * DINNER) return;
    int d = tid & 1023;
    int t = (tid >> 10) & 511;
    int b = tid >> 19;
    float acc = cb[d];
    #pragma unroll
    for (int j = 0; j < 4; ++j) {
        int tt = t - 3 + j;
        if (tt >= 0) acc += cw[d * 4 + j] * xz[(size_t)(b * NN + tt) * 2048 + d];
    }
    float r = acc / (1.f + expf(-acc));
    xc[tid] = r;
    xcb[tid] = (__bf16)r;
}

// ---------------------------------------------------------------------------
// Sequential selective scan (Heinsen recurrence of the reference's log-scan):
// h_t = (exp(dt*A)+eps) h_{t-1} + (|dt*B*xc|+eps);  y = sum_n h*C + xc*D
// 16 lanes = 16 states; 2 d-channels per wave. Output gated by silu(z), bf16.
// ---------------------------------------------------------------------------
__global__ __launch_bounds__(128)
void mamba_scan_kernel(const float* __restrict__ delta, const float* __restrict__ xdbl,
                       const float* __restrict__ xconv, const float* __restrict__ xz,
                       const float* __restrict__ A_log, const float* __restrict__ Dp,
                       __bf16* __restrict__ ygated)
{
    int wid  = (blockIdx.x * blockDim.x + threadIdx.x) >> 5;
    int lane = threadIdx.x & 31;
    if (wid >= BB * DINNER / 2) return;
    int half = lane >> 4;
    int n    = lane & 15;
    int bd   = wid * 2 + half;
    int b    = bd >> 10;
    int d    = bd & 1023;

    float Ac = -expf(A_log[d * DSTATE + n]);
    float Dc = Dp[d];
    float h  = 0.f;
    for (int t = 0; t < NN; ++t) {
        size_t row = (size_t)(b * NN + t);
        float dt = delta[row * DINNER + d];
        float xc = xconv[row * DINNER + d];
        float Bm = xdbl[row * 1056 + 1024 + n];
        float Cm = xdbl[row * 1056 + 1040 + n];
        float a  = expf(dt * Ac) + 1e-12f;
        float bb = fabsf(dt * Bm * xc) + 1e-12f;
        h = a * h + bb;
        float part = h * Cm;
        #pragma unroll
        for (int m = 8; m >= 1; m >>= 1) part += __shfl_xor(part, m, 16);
        if (n == 0) {
            float y = part + xc * Dc;
            float z = xz[row * 2048 + 1024 + d];
            float g = z / (1.f + expf(-z));
            ygated[row * DINNER + d] = (__bf16)(y * g);
        }
    }
}

// ---------------------------------------------------------------------------
extern "C" void kernel_launch(void* const* d_in, const int* in_sizes, int n_in,
                              void* d_out, int out_size, void* d_ws, size_t ws_size,
                              hipStream_t stream)
{
    (void)in_sizes; (void)n_in; (void)out_size; (void)ws_size;
    const float* x        = (const float*)d_in[0];
    const float* ctx      = (const float*)d_in[1];
    const float* Wq       = (const float*)d_in[2];
    const float* Wk       = (const float*)d_in[3];
    const float* Wv       = (const float*)d_in[4];
    const float* Wo_w     = (const float*)d_in[5];
    const float* Wo_b     = (const float*)d_in[6];
    const float* ss1_w    = (const float*)d_in[7];
    const float* ss1_b    = (const float*)d_in[8];
    const float* ss2_w    = (const float*)d_in[9];
    const float* ss2_b    = (const float*)d_in[10];
    const float* n1_w     = (const float*)d_in[11];
    const float* n1_b     = (const float*)d_in[12];
    const float* n2_w     = (const float*)d_in[13];
    const float* n2_b     = (const float*)d_in[14];
    const float* mbn_w    = (const float*)d_in[15];
    const float* mbn_b    = (const float*)d_in[16];
    const float* in_proj  = (const float*)d_in[17];
    const float* conv_w   = (const float*)d_in[18];
    const float* conv_b   = (const float*)d_in[19];
    const float* x_proj   = (const float*)d_in[20];
    const float* dt_w     = (const float*)d_in[21];
    const float* dt_b     = (const float*)d_in[22];
    const float* A_log    = (const float*)d_in[23];
    const float* Dp       = (const float*)d_in[24];
    const float* out_proj = (const float*)d_in[25];
    const float* lo_w     = (const float*)d_in[26];
    const float* lo_b     = (const float*)d_in[27];
    const float* scale1   = (const float*)d_in[28];
    const float* scale2   = (const float*)d_in[29];
    float* out = (float*)d_out;

    // ---- workspace carving (16B aligned) -------------------------------
    float* ws = (float*)d_ws;
    size_t off = 0;
    auto allocf = [&](size_t nel) { float* p = ws + off; off += (nel + 3) & ~(size_t)3; return p; };
    auto allocb = [&](size_t nel) { return (__bf16*)allocf((nel + 1) / 2); };

    // f32 buffers
    float* kbuf       = allocf((size_t)BB * MM * 512);
    float* vbuf       = allocf((size_t)BB * MM * 512);
    float* qbuf       = allocf((size_t)BB * NN * 512);
    float* xattn_raw  = allocf((size_t)BB * NN * 512);
    float* xattn_buf  = allocf((size_t)BB * NN * 512);
    float* smean      = allocf((size_t)BB * CTXD);
    float* ss1        = allocf((size_t)BB * 1024);
    float* ss2        = allocf((size_t)BB * 1024);
    float* xz         = allocf((size_t)BB * NN * 2048);
    float* xconv      = allocf((size_t)BB * NN * DINNER);
    float* xdbl       = allocf((size_t)BB * NN * 1056);
    float* delta      = allocf((size_t)BB * NN * DINNER);
    float* xmamba_raw = allocf((size_t)BB * NN * 512);
    // transposed bf16 weights [N x K]
    __bf16* WqT  = allocb((size_t)512 * 512);
    __bf16* WkT  = allocb((size_t)512 * CTXD);
    __bf16* WvT  = allocb((size_t)512 * CTXD);
    __bf16* WoT  = allocb((size_t)512 * 512);
    __bf16* ipT  = allocb((size_t)2048 * 512);
    __bf16* xpT  = allocb((size_t)1056 * 1024);
    __bf16* dtT  = allocb((size_t)1024 * 1024);
    __bf16* opT  = allocb((size_t)512 * 1024);
    __bf16* loT  = allocb((size_t)512 * 512);
    // bf16 activation (A-operand) buffers
    __bf16* ctx_bf    = allocb((size_t)160 * CTXD);       // 154 rows padded to 160
    __bf16* x_bf      = allocb((size_t)BB * NN * 512);
    __bf16* attn_bf   = allocb((size_t)BB * NN * 512);
    __bf16* xn_bf     = allocb((size_t)BB * NN * 512);
    __bf16* xconv_bf  = allocb((size_t)BB * NN * DINNER);
    __bf16* xdbl_bf   = allocb((size_t)BB * NN * 1056);
    __bf16* ygated_bf = allocb((size_t)BB * NN * DINNER);
    __bf16* y2_bf     = allocb((size_t)BB * NN * 512);

    const int ROWS = BB * NN;   // 1024

    auto wt = [&](const float* W, __bf16* Wt, int K, int N) {
        wt_cvt_kernel<<<(K * N + 255) / 256, 256, 0, stream>>>(W, Wt, K, N);
    };
    auto gemm_tiles = [&](int M, int N, int NT) {
        return (((M + 31) >> 5) * (N / (16 * NT)) + 3) >> 2;
    };

    // ---- weight conversions --------------------------------------------
    wt(Wq, WqT, 512, 512);      wt(Wk, WkT, CTXD, 512);   wt(Wv, WvT, CTXD, 512);
    wt(Wo_w, WoT, 512, 512);    wt(in_proj, ipT, 512, 2048);
    wt(x_proj, xpT, 1024, 1056); wt(dt_w, dtT, 1024, 1024);
    wt(out_proj, opT, 1024, 512); wt(lo_w, loT, 512, 512);

    // ---- activation conversions ----------------------------------------
    cvt_pad_kernel<<<(160 * CTXD + 255) / 256, 256, 0, stream>>>(ctx, ctx_bf, BB * MM, 160, CTXD);
    cvt_pad_kernel<<<(ROWS * 512 + 255) / 256, 256, 0, stream>>>(x, x_bf, ROWS, ROWS, 512);

    // ---- context-conditioned scale/shift -------------------------------
    ctx_mean_silu_kernel<<<(BB * CTXD + 255) / 256, 256, 0, stream>>>(ctx, smean);
    ss_proj_kernel<<<(BB * 1024 + 255) / 256, 256, 0, stream>>>(smean, ss1_w, ss1_b, ss1);
    ss_proj_kernel<<<(BB * 1024 + 255) / 256, 256, 0, stream>>>(smean, ss2_w, ss2_b, ss2);

    // ---- attention branch ----------------------------------------------
    gemm_bf16_kernel<0,4,0><<<gemm_tiles(BB*MM,512,4), 128, 0, stream>>>(
        ctx_bf, CTXD, WkT, nullptr, kbuf, nullptr, 512, BB*MM, 512, CTXD);
    gemm_bf16_kernel<0,4,0><<<gemm_tiles(BB*MM,512,4), 128, 0, stream>>>(
        ctx_bf, CTXD, WvT, nullptr, vbuf, nullptr, 512, BB*MM, 512, CTXD);
    gemm_bf16_kernel<0,4,0><<<gemm_tiles(ROWS,512,4), 128, 0, stream>>>(
        x_bf, 512, WqT, nullptr, qbuf, nullptr, 512, ROWS, 512, 512);
    attention_kernel<<<(BB * 8 * NN * 32 + 127) / 128, 128, 0, stream>>>(qbuf, kbuf, vbuf, attn_bf);
    gemm_bf16_kernel<1,4,0><<<gemm_tiles(ROWS,512,4), 128, 0, stream>>>(
        attn_bf, 512, WoT, Wo_b, xattn_raw, nullptr, 512, ROWS, 512, 512);
    post_ss_ln_kernel<<<(ROWS * 32 + 127) / 128, 128, 0, stream>>>(
        xattn_raw, ss1, n1_w, n1_b, scale1, nullptr, nullptr, xattn_buf, ROWS);

    // ---- mamba branch ---------------------------------------------------
    layernorm_rows_kernel<<<(ROWS * 32 + 127) / 128, 128, 0, stream>>>(x, mbn_w, mbn_b, xn_bf, ROWS);
    gemm_bf16_kernel<0,4,0><<<gemm_tiles(ROWS,2048,4), 128, 0, stream>>>(
        xn_bf, 512, ipT, nullptr, xz, nullptr, 2048, ROWS, 2048, 512);
    conv_silu_kernel<<<(ROWS * DINNER + 255) / 256, 256, 0, stream>>>(xz, conv_w, conv_b, xconv, xconv_bf);
    gemm_bf16_kernel<0,2,2><<<gemm_tiles(ROWS,1056,2), 128, 0, stream>>>(
        xconv_bf, DINNER, xpT, nullptr, xdbl, xdbl_bf, 1056, ROWS, 1056, DINNER);
    gemm_bf16_kernel<2,4,0><<<gemm_tiles(ROWS,1024,4), 128, 0, stream>>>(
        xdbl_bf, 1056, dtT, dt_b, delta, nullptr, 1024, ROWS, 1024, 1024);
    mamba_scan_kernel<<<((BB * DINNER / 2) * 32 + 127) / 128, 128, 0, stream>>>(
        delta, xdbl, xconv, xz, A_log, Dp, ygated_bf);
    gemm_bf16_kernel<0,4,1><<<gemm_tiles(ROWS,512,4), 128, 0, stream>>>(
        ygated_bf, DINNER, opT, nullptr, nullptr, y2_bf, 512, ROWS, 512, DINNER);
    gemm_bf16_kernel<1,4,0><<<gemm_tiles(ROWS,512,4), 128, 0, stream>>>(
        y2_bf, 512, loT, lo_b, xmamba_raw, nullptr, 512, ROWS, 512, 512);

    // ---- final combine ---------------------------------------------------
    post_ss_ln_kernel<<<(ROWS * 32 + 127) / 128, 128, 0, stream>>>(
        xmamba_raw, ss2, n2_w, n2_b, scale2, x, xattn_buf, out, ROWS);
}
